// EdgeConvBlock_3882650436819
// MI455X (gfx1250) — compile-verified, matrix operand order checked by reading, not run on hardware
//
#include <hip/hip_runtime.h>

#define B_    8
#define N_    2048
#define KNN_  32
#define COLS_ 128   // 4 points * 32 neighbors per workgroup
#define TPB_  256
#define NT_   64    // neighbor-tile size staged in LDS by the TDM

typedef __attribute__((ext_vector_type(16))) __bf16          v16bf;
typedef __attribute__((ext_vector_type(16))) unsigned short  v16us;
typedef __attribute__((ext_vector_type(8)))  unsigned short  v8us;
typedef __attribute__((ext_vector_type(8)))  float           v8f;
typedef __attribute__((ext_vector_type(4)))  unsigned int    u32x4;
typedef __attribute__((ext_vector_type(8)))  int             i32x8;
typedef __attribute__((ext_vector_type(4)))  int             i32x4;

struct us16pair { v8us lo, hi; };   // 32B, bit_cast'able to v16bf

__device__ __forceinline__ unsigned short f2bf(float f) {
  unsigned u = __builtin_bit_cast(unsigned, f);
  u += 0x7FFFu + ((u >> 16) & 1u);          // round-to-nearest-even
  return (unsigned short)(u >> 16);
}

// K index for element e (0..15) of a 16-bit 16x32 WMMA A/B fragment (ISA 7.12.2)
__device__ __forceinline__ int kmap(int e, int half) {
  int p = e >> 1;
  return ((p >> 2) << 4) + (half << 3) + ((p & 3) << 1) + (e & 1);
}

// Inverse mapping: (k, column j) -> ushort index into the packed fragment pool
// pool layout: [kstep][ntile(8)][lane(32)][elem(16)], 32B per lane-fragment
__device__ __forceinline__ int packAddr(int k, int j) {
  int ks = k >> 5, K = k & 31;
  int e    = (((((K >> 4) << 2) | ((K >> 1) & 3)) << 1) | (K & 1));
  int lane = (j & 15) + (((K >> 3) & 1) << 4);
  return ((ks * 8 + (j >> 4)) * 32 + lane) * 16 + e;
}

__device__ __forceinline__ v16bf loadFrag(const unsigned short* p) {
  us16pair f;
  f.lo = *(const v8us*)p;        // ds_load_b128
  f.hi = *(const v8us*)(p + 8);  // ds_load_b128
  return __builtin_bit_cast(v16bf, f);
}

// TDM: DMA a 2D tile (rows x NT_ f32, row stride N_) from global into LDS.
// D# per CDNA5 ISA ch.8: group0 {count, lds_addr, global_addr, type=2},
// group1 {data_size=4B, tensor_dim0/1, tile_dim0/1, tensor_dim0_stride}.
__device__ __forceinline__ void tdm_load_tile(const float* gsrc, void* lds_dst,
                                              int rows) {
  unsigned long long ga = (unsigned long long)(const void*)gsrc;
  unsigned lds_off = (unsigned)(size_t)lds_dst;
  u32x4 g0;
  g0[0] = 1u;                                   // count=1, user descriptor
  g0[1] = lds_off;                              // lds_addr
  g0[2] = (unsigned)ga;                         // global_addr[31:0]
  g0[3] = (unsigned)(ga >> 32) | (2u << 30);    // global_addr[56:32] | type=2
  i32x8 g1;
  g1[0] = 2 << 16;                              // data_size code 2 = 4 bytes
  g1[1] = NT_ << 16;                            // tensor_dim0[15:0] = NT_
  g1[2] = rows << 16;                           // dim0 hi=0 | tensor_dim1 = rows
  g1[3] = NT_ << 16;                            // dim1 hi=0 | tile_dim0 = NT_
  g1[4] = rows;                                 // tile_dim1 = rows, tile_dim2=0
  g1[5] = N_;                                   // tensor_dim0_stride[31:0] = N_
  g1[6] = 0;                                    // stride0 hi | stride1 lo
  g1[7] = 0;
  i32x4 z4 = {0, 0, 0, 0};                      // groups 2/3 unused (2D tensor)
  i32x8 z8 = {0, 0, 0, 0, 0, 0, 0, 0};
  __builtin_amdgcn_tensor_load_to_lds(g0, g1, z4, z4, z8, 0);
}

// ---------------------------------------------------------------------------
// kNN: one wave per query point; neighbor features staged in LDS tiles by the
// Tensor Data Mover and shared by all 4 waves (4x less L2 traffic).
// Distance row lives in LDS; lane L owns slots n with n%32==L, so the 32
// extraction passes have no cross-lane LDS hazards.
// ---------------------------------------------------------------------------
template <int C>
__global__ __launch_bounds__(128) void knn_kernel(const float* __restrict__ x,
                                                  int* __restrict__ idx_out) {
  __shared__ float qfeat[4][C];
  __shared__ float dist[4][N_];
  __shared__ __align__(16) float ntile[C][NT_];
  const int wave = threadIdx.x >> 5;
  const int lane = threadIdx.x & 31;
  const int g = blockIdx.x * 4 + wave;
  const int b = g >> 11;            // N_ == 2048; all 4 queries share b
  const int q = g & (N_ - 1);
  const float* xb = x + (size_t)b * C * N_;

  for (int c = lane; c < C; c += 32) qfeat[wave][c] = xb[c * N_ + q];
  __syncthreads();

  float xxq = 0.f;
#pragma unroll
  for (int c = 0; c < C; ++c) { float t = qfeat[wave][c]; xxq += t * t; }

  for (int tile = 0; tile < N_ / NT_; ++tile) {
    const int nbase = tile * NT_;
    if (wave == 0) {                              // one TDM op per tile
      tdm_load_tile(xb + nbase, &ntile[0][0], C);
      __builtin_amdgcn_s_wait_tensorcnt(0);
    }
    __syncthreads();                              // tile visible to all waves
    for (int t = lane; t < NT_; t += 32) {
      float dot = 0.f, xxn = 0.f;
#pragma unroll
      for (int c = 0; c < C; ++c) {
        float xn = ntile[c][t];
        dot += qfeat[wave][c] * xn;
        xxn += xn * xn;
      }
      dist[wave][nbase + t] = 2.f * dot - xxq - xxn;  // neg squared distance
    }
    __syncthreads();                              // done before next DMA
  }

  int* row_out = idx_out + ((size_t)b * N_ + q) * KNN_;
  for (int kk = 0; kk < KNN_; ++kk) {
    float v = -3.4e38f; int id = 0x7FFFFFFF;
    for (int n = lane; n < N_; n += 32) {
      float d = dist[wave][n];
      if (d > v || (d == v && n < id)) { v = d; id = n; }
    }
#pragma unroll
    for (int off = 16; off >= 1; off >>= 1) {
      float ov = __shfl_xor(v, off, 32);
      int   oi = __shfl_xor(id, off, 32);
      if (ov > v || (ov == v && oi < id)) { v = ov; id = oi; }
    }
    if (lane == 0) row_out[kk] = id;
    if ((id & 31) == lane) dist[wave][id] = -3.4e38f;  // owner lane retires it
  }
}

// ---------------------------------------------------------------------------
// Fused EdgeConv block: gather edges (packed bf16 fragments in LDS) ->
// WMMA GEMM1 -> BN+LReLU (packed y1) -> WMMA GEMM2 -> BN+LReLU -> neighbor max
// ---------------------------------------------------------------------------
template <int CIN, int K1>
__global__ __launch_bounds__(TPB_) void edgeconv_kernel(
    const float* __restrict__ x,     // [B, CIN, N]
    const int*   __restrict__ idx,   // [B, N, 32]
    const float* __restrict__ w1,    // [64, 2*CIN]
    const float* __restrict__ g1, const float* __restrict__ bb1,
    const float* __restrict__ m1, const float* __restrict__ v1,
    const float* __restrict__ w2,    // [64, 64]
    const float* __restrict__ g2, const float* __restrict__ bb2,
    const float* __restrict__ m2, const float* __restrict__ v2,
    float* __restrict__ out,         // [B, 192, N]
    float* __restrict__ xl,          // [B, 64, N]
    int layer) {
  constexpr int KD  = 2 * CIN;                     // real K of GEMM1
  constexpr int KS1 = K1 / 32;                     // GEMM1 k-steps
  // ePack (bf16 fragment pool, K1*COLS us) aliased with y2 (f32 [64][COLS])
  __shared__ __align__(32) char ebuf[64 * COLS_ * 4];          // 32 KB
  __shared__ __align__(32) unsigned short y1Pack[64 * COLS_];  // 16 KB
  __shared__ int   nbrIdx[COLS_];
  __shared__ float sc1[64], sh1[64], sc2[64], sh2[64];

  unsigned short* ePack = (unsigned short*)ebuf;
  float*          y2    = (float*)ebuf;

  const int tid  = threadIdx.x;
  const int lane = tid & 31;
  const int wave = tid >> 5;
  const int half = lane >> 4;
  const int ln   = lane & 15;
  const int b  = blockIdx.x / (N_ / 4);
  const int n0 = (blockIdx.x % (N_ / 4)) * 4;

  for (int c = tid; c < 64; c += TPB_) {
    float s1 = g1[c] * rsqrtf(v1[c] + 1e-5f);
    sc1[c] = s1; sh1[c] = bb1[c] - m1[c] * s1;
    float s2 = g2[c] * rsqrtf(v2[c] + 1e-5f);
    sc2[c] = s2; sh2[c] = bb2[c] - m2[c] * s2;
  }
  for (int j = tid; j < COLS_; j += TPB_)
    nbrIdx[j] = idx[((size_t)b * N_ + n0 + (j >> 5)) * KNN_ + (j & 31)];
  if (K1 > KD) {                      // zero-pad fragment pool (layer 0 only)
    for (int t = tid; t < K1 * COLS_; t += TPB_) ePack[t] = 0;
  }
  __syncthreads();

  // Build edge features straight into the packed fragment pool
  const float* xb = x + (size_t)b * CIN * N_;
  for (int t = tid; t < CIN * COLS_; t += TPB_) {
    int c = t / COLS_, j = t % COLS_;
    int n = n0 + (j >> 5);
    float ctr = xb[c * N_ + n];
    float nb  = xb[c * N_ + nbrIdx[j]];
    ePack[packAddr(c, j)]       = f2bf(ctr);
    ePack[packAddr(CIN + c, j)] = f2bf(nb - ctr);
  }
  __syncthreads();

  // ---- GEMM1: each wave owns m-tile (wave&3); A fragments live in registers
  const int mt = wave & 3, m0 = mt * 16;
  {
    v16us a1[KS1];
#pragma unroll
    for (int ks = 0; ks < KS1; ++ks)
#pragma unroll
      for (int e = 0; e < 16; ++e) {
        int kk = ks * 32 + kmap(e, half);
        a1[ks][e] = (kk < KD) ? f2bf(w1[(m0 + ln) * KD + kk]) : (unsigned short)0;
      }
    for (int nt = (wave >> 2); nt < 8; nt += 2) {
      v8f acc = {};
#pragma unroll
      for (int ks = 0; ks < KS1; ++ks) {
        v16bf bfr = loadFrag(ePack + ((ks * 8 + nt) * 32 + lane) * 16);
        acc = __builtin_amdgcn_wmma_f32_16x16x32_bf16(
            false, __builtin_bit_cast(v16bf, a1[ks]), false, bfr,
            (short)0, acc, false, false);
      }
#pragma unroll
      for (int r = 0; r < 8; ++r) {
        int ch = m0 + r + (half << 3);
        int col = nt * 16 + ln;
        float vv = acc[r] * sc1[ch] + sh1[ch];
        vv = vv > 0.f ? vv : 0.2f * vv;
        y1Pack[packAddr(ch, col)] = f2bf(vv);   // packed for GEMM2's B operand
      }
    }
  }
  __syncthreads();

  // ---- GEMM2: W2[64x64] * y1[64xCOLS]; writes f32 y2 (aliases ePack)
  {
    v16us a2[2];
#pragma unroll
    for (int ks = 0; ks < 2; ++ks)
#pragma unroll
      for (int e = 0; e < 16; ++e)
        a2[ks][e] = f2bf(w2[(m0 + ln) * 64 + ks * 32 + kmap(e, half)]);
    for (int nt = (wave >> 2); nt < 8; nt += 2) {
      v8f acc = {};
#pragma unroll
      for (int ks = 0; ks < 2; ++ks) {
        v16bf bfr = loadFrag(y1Pack + ((ks * 8 + nt) * 32 + lane) * 16);
        acc = __builtin_amdgcn_wmma_f32_16x16x32_bf16(
            false, __builtin_bit_cast(v16bf, a2[ks]), false, bfr,
            (short)0, acc, false, false);
      }
#pragma unroll
      for (int r = 0; r < 8; ++r) {
        int ch = m0 + r + (half << 3);
        int col = nt * 16 + ln;
        float vv = acc[r] * sc2[ch] + sh2[ch];
        y2[ch * COLS_ + col] = vv > 0.f ? vv : 0.2f * vv;
      }
    }
  }
  __syncthreads();

  // ---- Max over the 32 neighbor columns; write x_l and the concat slot
  for (int t = tid; t < 64 * 4; t += TPB_) {
    int ch = t & 63, p = t >> 6;
    float m = -3.4e38f;
#pragma unroll
    for (int kk = 0; kk < 32; ++kk)
      m = fmaxf(m, y2[ch * COLS_ + p * 32 + kk]);
    int n = n0 + p;
    xl [((size_t)b * 64 + ch) * N_ + n] = m;
    out[((size_t)b * 192 + 64 * layer + ch) * N_ + n] = m;
  }
}

// ---------------------------------------------------------------------------
extern "C" void kernel_launch(void* const* d_in, const int* in_sizes, int n_in,
                              void* d_out, int out_size, void* d_ws, size_t ws_size,
                              hipStream_t stream) {
  (void)in_sizes; (void)n_in; (void)out_size; (void)ws_size;
  const float* x = (const float*)d_in[0];
  const float* p0[10]; const float* p1[10]; const float* p2[10];
  for (int i = 0; i < 10; ++i) {
    p0[i] = (const float*)d_in[1 + i];
    p1[i] = (const float*)d_in[11 + i];
    p2[i] = (const float*)d_in[21 + i];
  }
  float* out = (float*)d_out;
  float* x0 = out + (size_t)B_ * 192 * N_;
  float* x1 = x0 + (size_t)B_ * 64 * N_;
  float* x2 = x1 + (size_t)B_ * 64 * N_;
  int* idx = (int*)d_ws;   // [B, N, 32] = 2 MB

  dim3 blkK(128), grdK(B_ * N_ / 4);
  dim3 blkE(TPB_), grdE(B_ * (N_ / 4));

  // Layer 0: CIN=3, GEMM1 K padded 6 -> 32
  knn_kernel<3><<<grdK, blkK, 0, stream>>>(x, idx);
  edgeconv_kernel<3, 32><<<grdE, blkE, 0, stream>>>(x, idx,
      p0[0], p0[1], p0[2], p0[3], p0[4], p0[5], p0[6], p0[7], p0[8], p0[9],
      out, x0, 0);
  // Layer 1: CIN=64, K=128
  knn_kernel<64><<<grdK, blkK, 0, stream>>>(x0, idx);
  edgeconv_kernel<64, 128><<<grdE, blkE, 0, stream>>>(x0, idx,
      p1[0], p1[1], p1[2], p1[3], p1[4], p1[5], p1[6], p1[7], p1[8], p1[9],
      out, x1, 1);
  // Layer 2: CIN=64, K=128
  knn_kernel<64><<<grdK, blkK, 0, stream>>>(x1, idx);
  edgeconv_kernel<64, 128><<<grdE, blkE, 0, stream>>>(x1, idx,
      p2[0], p2[1], p2[2], p2[3], p2[4], p2[5], p2[6], p2[7], p2[8], p2[9],
      out, x2, 2);
}
